// SequentialRNN_36764920054106
// MI455X (gfx1250) — matrix operand order repaired and checked by reading, not compiled, not run on hardware
//
#include <hip/hip_runtime.h>
#include <hip/hip_bf16.h>
#include <math.h>

// ---------------------------------------------------------------------------
// Fused 2-layer Elman RNN + FC for MI455X (gfx1250, wave32, WMMA bf16).
//   layer1: h1_t = tanh(x_t W_ih1^T + (b_ih1+b_hh1) + h1_{t-1} W_hh1^T)
//   layer2: h2_t = tanh(h1_t W_ih2^T + (b_ih2+b_hh2) + h2_{t-1} W_hh2^T)
//   out    = concat_t(h2_t) @ W_fc^T + b_fc
// One wave owns 16 batch rows; h-states live in LDS as bf16; all GEMMs are
// v_wmma_f32_16x16x32_bf16 tiles.  Weights are pre-swizzled into
// fragment-major LDS so every B-fragment is one contiguous 32-byte LDS load;
// A-fragments are two contiguous 16-byte runs.  Activations use the native
// gfx1250 v_tanh_f32 when the builtin is available.
// ---------------------------------------------------------------------------

#define BATCH 16384
#define T_SEQ 28
#define I_IN  28             // padded to 32 in the K dimension
#define H1    128
#define H2    64
#define NCLS  10
#define FDIM  (H2 * T_SEQ)   // 1792
#define WAVES 8
#define BLK   (WAVES * 32)   // 256 threads
#define ROWS_PER_WAVE 16
#define ROWS_PER_BLK  (WAVES * ROWS_PER_WAVE)  // 128
#define XSTRIDE (T_SEQ * I_IN)                 // 784 floats per batch row

typedef __attribute__((ext_vector_type(16))) __bf16 v16bf;
typedef __attribute__((ext_vector_type(8)))  __bf16 v8bf;
typedef __attribute__((ext_vector_type(8)))  float  v8f;

static __device__ __forceinline__ v8f wmma_bf16(v16bf a, v16bf b, v8f c) {
  // 8 args: (neg_a, A, neg_b, B, c_mod, C, reuse_a, reuse_b)
  return __builtin_amdgcn_wmma_f32_16x16x32_bf16(false, a, false, b,
                                                 (short)0, c, false, false);
}

// Activation: native v_tanh_f32 on gfx1250 when clang exposes it; otherwise
// clamped (e^{2x}-1)/(e^{2x}+1) with v_exp_f32 + v_rcp_f32.
static __device__ __forceinline__ float fast_tanh(float x) {
#if __has_builtin(__builtin_amdgcn_tanhf)
  return __builtin_amdgcn_tanhf(x);
#elif __has_builtin(__builtin_amdgcn_tanh_f32)
  return __builtin_amdgcn_tanh_f32(x);
#else
  x = fminf(fmaxf(x, -15.0f), 15.0f);          // keep exp finite
  const float e = __expf(2.0f * x);
  return (e - 1.0f) * __builtin_amdgcn_rcpf(e + 1.0f);
#endif
}

// ---------------------------------------------------------------------------
// Fragment loaders.
// A-matrix (16x32, 16-bit) per CDNA5 ISA 7.12.2: lane l -> M = l&15,
// khi = l>>4; halves 0..7 hold K = khi*8..+7, halves 8..15 hold K = 16+khi*8..+7
// (two contiguous runs).  B-matrix (32x16): lane l -> N = l&15, half i ->
// K = (l>>4)*8 + (i&7) + (i>>3)*16, as for A.  C/D: lane l -> N = l&15, VGPR r -> M = r + 8*(l>>4).
// ---------------------------------------------------------------------------

// A fragment from a row-major bf16 LDS matrix (two ds_load_b128 per lane).
static __device__ __forceinline__ v16bf lds_loadA(const __bf16* base, int ld, int lane) {
  const int m = lane & 15, khi = lane >> 4;
  const __bf16* p = base + m * ld + khi * 8;
  const v8bf lo = *(const v8bf*)(p);
  const v8bf hi = *(const v8bf*)(p + 16);
  return __builtin_shufflevector(lo, hi, 0, 1, 2, 3, 4, 5, 6, 7,
                                 8, 9, 10, 11, 12, 13, 14, 15);
}

// B fragment from the fragment-major swizzled weight cache: each lane's 16
// halves are contiguous -> one 32B LDS vector load.
static __device__ __forceinline__ v16bf lds_loadBfrag(const __bf16* frags,
                                                      int tile, int lane) {
  return *(const v16bf*)(frags + (tile << 9) + (lane << 4));
}

// A fragment for x_t straight from HBM: 4 aligned float4 loads, branchless;
// the khi==1 tail quad (k=28..31) is a safe dummy load zeroed by selects.
static __device__ __forceinline__ v16bf glb_loadXA(const float* xr, int lane) {
  const int m = lane & 15, khi = lane >> 4;
  const float* pm = xr + (size_t)m * XSTRIDE;
  const int k0 = khi * 8;
  const float4 f0 = *(const float4*)(pm + k0);
  const float4 f1 = *(const float4*)(pm + k0 + 4);
  const float4 f2 = *(const float4*)(pm + k0 + 16);
  const float4 f3 = *(const float4*)(pm + 20);    // real for khi=0, dummy for khi=1
  const bool hi = (khi != 0);
  v16bf a;
  a[0]  = (__bf16)f0.x;  a[1]  = (__bf16)f0.y;
  a[2]  = (__bf16)f0.z;  a[3]  = (__bf16)f0.w;
  a[4]  = (__bf16)f1.x;  a[5]  = (__bf16)f1.y;
  a[6]  = (__bf16)f1.z;  a[7]  = (__bf16)f1.w;
  a[8]  = (__bf16)f2.x;  a[9]  = (__bf16)f2.y;
  a[10] = (__bf16)f2.z;  a[11] = (__bf16)f2.w;
  a[12] = hi ? (__bf16)0.0f : (__bf16)f3.x;
  a[13] = hi ? (__bf16)0.0f : (__bf16)f3.y;
  a[14] = hi ? (__bf16)0.0f : (__bf16)f3.z;
  a[15] = hi ? (__bf16)0.0f : (__bf16)f3.w;
  return a;
}

// B fragment of W_fc (rows n<10 valid): clamped row + 4 float4 loads + selects.
static __device__ __forceinline__ v16bf glb_loadWfcB(const float* wfc, int f0, int lane) {
  const int n = lane & 15, khi = lane >> 4;
  const int nr = (n < NCLS) ? n : 0;
  const bool valid = (n < NCLS);
  const float* p = wfc + (size_t)nr * FDIM + f0 + khi * 8;
  const float4 q0 = *(const float4*)(p);
  const float4 q1 = *(const float4*)(p + 4);
  const float4 q2 = *(const float4*)(p + 16);
  const float4 q3 = *(const float4*)(p + 20);
  v16bf b;
  b[0]  = valid ? (__bf16)q0.x : (__bf16)0.0f;
  b[1]  = valid ? (__bf16)q0.y : (__bf16)0.0f;
  b[2]  = valid ? (__bf16)q0.z : (__bf16)0.0f;
  b[3]  = valid ? (__bf16)q0.w : (__bf16)0.0f;
  b[4]  = valid ? (__bf16)q1.x : (__bf16)0.0f;
  b[5]  = valid ? (__bf16)q1.y : (__bf16)0.0f;
  b[6]  = valid ? (__bf16)q1.z : (__bf16)0.0f;
  b[7]  = valid ? (__bf16)q1.w : (__bf16)0.0f;
  b[8]  = valid ? (__bf16)q2.x : (__bf16)0.0f;
  b[9]  = valid ? (__bf16)q2.y : (__bf16)0.0f;
  b[10] = valid ? (__bf16)q2.z : (__bf16)0.0f;
  b[11] = valid ? (__bf16)q2.w : (__bf16)0.0f;
  b[12] = valid ? (__bf16)q3.x : (__bf16)0.0f;
  b[13] = valid ? (__bf16)q3.y : (__bf16)0.0f;
  b[14] = valid ? (__bf16)q3.z : (__bf16)0.0f;
  b[15] = valid ? (__bf16)q3.w : (__bf16)0.0f;
  return b;
}

// tanh + bias the 16x16 f32 accumulator, store bf16 into the LDS h-state.
static __device__ __forceinline__ void store_tanh(__bf16* h, int ld, int n0,
                                                  v8f acc, const float* bias,
                                                  int lane) {
  const int n = lane & 15, khi = lane >> 4;
  const float bb = bias[n0 + n];
#pragma unroll
  for (int r = 0; r < 8; ++r)
    h[(r + 8 * khi) * ld + n0 + n] = (__bf16)fast_tanh(acc[r] + bb);
}

// Stage a row-major fp32 weight [Nrows x Ksrc] as bf16 B-fragments in LDS,
// K padded to KC*32.  Fragment idx -> (tile = kc*NT+nt, lane, half i) with
// k = kc*32 + (lane>>4)*8 + (i&7) + (i>>3)*16, n = nt*16 + (lane&15).
static __device__ __forceinline__ void stage_frags(__bf16* dst, const float* W,
                                                   int NT, int KC, int Ksrc,
                                                   int tid) {
  const int total = KC * NT * 512;
  for (int idx = tid; idx < total; idx += BLK) {
    const int i    = idx & 15;
    const int lane = (idx >> 4) & 31;
    const int tile = idx >> 9;
    const int kc = tile / NT, nt = tile - kc * NT;   // NT is a power of two
    const int k = kc * 32 + ((lane >> 4) << 3) + (i & 7) + ((i >> 3) << 4);
    const int n = nt * 16 + (lane & 15);
    dst[idx] = (k < Ksrc) ? (__bf16)W[(size_t)n * Ksrc + k] : (__bf16)0.0f;
  }
}

__global__ __launch_bounds__(BLK)
void rnn2_fused_wmma(const float* __restrict__ x,
                     const float* __restrict__ W_ih1, const float* __restrict__ W_hh1,
                     const float* __restrict__ b_ih1, const float* __restrict__ b_hh1,
                     const float* __restrict__ W_ih2, const float* __restrict__ W_hh2,
                     const float* __restrict__ b_ih2, const float* __restrict__ b_hh2,
                     const float* __restrict__ W_fc,  const float* __restrict__ b_fc,
                     float* __restrict__ out) {
  extern __shared__ __align__(32) char smem[];
  // Fragment-major bf16 weight caches (shared by the whole block):
  __bf16* wih1 = (__bf16*)smem;              // 1 kc x 8 nt tiles  ( 4096 h)
  __bf16* whh1 = wih1 + 1 * 8 * 512;         // 4 kc x 8 nt tiles  (16384 h)
  __bf16* wih2 = whh1 + 4 * 8 * 512;         // 4 kc x 4 nt tiles  ( 8192 h)
  __bf16* whh2 = wih2 + 4 * 4 * 512;         // 2 kc x 4 nt tiles  ( 4096 h)
  __bf16* h1all = whh2 + 2 * 4 * 512;        // WAVES x 16 x 128 state
  __bf16* h2all = h1all + WAVES * 16 * H1;   // WAVES x 16 x  64 state
  float*  bias1 = (float*)(h2all + WAVES * 16 * H2);   // 128
  float*  bias2 = bias1 + H1;                          //  64

  const int tid  = threadIdx.x;
  const int lane = tid & 31;
  const int wave = tid >> 5;

  // ---- stage weights as pre-swizzled B-fragments (bf16, K zero-padded) ----
  stage_frags(wih1, W_ih1, 8, 1, I_IN, tid);   // K: 28 -> 32
  stage_frags(whh1, W_hh1, 8, 4, H1,   tid);
  stage_frags(wih2, W_ih2, 4, 4, H1,   tid);
  stage_frags(whh2, W_hh2, 4, 2, H2,   tid);
  for (int n = tid; n < H1; n += BLK) bias1[n] = b_ih1[n] + b_hh1[n];
  for (int n = tid; n < H2; n += BLK) bias2[n] = b_ih2[n] + b_hh2[n];
  for (int i = tid; i < WAVES * 16 * (H1 + H2); i += BLK)
    h1all[i] = (__bf16)0.0f;   // h1all & h2all contiguous; h0 = 0
  __syncthreads();
  // Everything below is wave-private: no barriers inside the T loop.

  __bf16* h1w = h1all + wave * 16 * H1;      // row-major [16][128]
  __bf16* h2w = h2all + wave * 16 * H2;      // row-major [16][64]
  const int rowBase = (blockIdx.x * WAVES + wave) * ROWS_PER_WAVE;
  const float* xw = x + (size_t)rowBase * XSTRIDE;

  v8f accFC = {};   // persistent FC accumulator (N=10 padded to 16)

  for (int t = 0; t < T_SEQ; ++t) {
    const float* xr = xw + t * I_IN;
    if (t + 1 < T_SEQ)
      __builtin_prefetch(xr + I_IN, 0, 0);   // pull next step's x rows

    // ---------------- layer 1: [16x128] = x_t[16x32] + h1[16x128] chain ----
    const v16bf ax = glb_loadXA(xr, lane);
    v16bf ah1[4];
#pragma unroll
    for (int kc = 0; kc < 4; ++kc)
      ah1[kc] = lds_loadA(h1w + kc * 32, H1, lane);

    v8f acc1[8];
#pragma unroll
    for (int nt = 0; nt < 8; ++nt) {
      v8f c = {};
      c = wmma_bf16(ax, lds_loadBfrag(wih1, nt, lane), c);
#pragma unroll
      for (int kc = 0; kc < 4; ++kc)
        c = wmma_bf16(ah1[kc], lds_loadBfrag(whh1, kc * 8 + nt, lane), c);
      acc1[nt] = c;
    }
#pragma unroll
    for (int nt = 0; nt < 8; ++nt)
      store_tanh(h1w, H1, nt * 16, acc1[nt], bias1, lane);

    // ---------------- layer 2: [16x64] = h1_t[16x128] + h2[16x64] chain ----
    v16bf a12[4];
#pragma unroll
    for (int kc = 0; kc < 4; ++kc)
      a12[kc] = lds_loadA(h1w + kc * 32, H1, lane);   // freshly written h1_t
    v16bf ah2[2];
#pragma unroll
    for (int kc = 0; kc < 2; ++kc)
      ah2[kc] = lds_loadA(h2w + kc * 32, H2, lane);   // h2_{t-1}

#pragma unroll
    for (int nt = 0; nt < 4; ++nt) {
      v8f c = {};
#pragma unroll
      for (int kc = 0; kc < 4; ++kc)
        c = wmma_bf16(a12[kc], lds_loadBfrag(wih2, kc * 4 + nt, lane), c);
#pragma unroll
      for (int kc = 0; kc < 2; ++kc)
        c = wmma_bf16(ah2[kc], lds_loadBfrag(whh2, kc * 4 + nt, lane), c);
      store_tanh(h2w, H2, nt * 16, c, bias2, lane);
    }

    // ---------------- FC accumulation: out += h2_t @ W_fc[:, t*64:+64]^T ---
#pragma unroll
    for (int kc = 0; kc < 2; ++kc) {
      const v16bf af = lds_loadA(h2w + kc * 32, H2, lane);
      const v16bf bf = glb_loadWfcB(W_fc, t * H2 + kc * 32, lane);
      accFC = wmma_bf16(af, bf, accFC);
    }
  }

  // ---- epilogue: add b_fc, write the wave's [16 x 10] output block --------
  const int n = lane & 15, khi = lane >> 4;
  if (n < NCLS) {
    const float bb = b_fc[n];
#pragma unroll
    for (int r = 0; r < 8; ++r)
      out[(size_t)(rowBase + r + 8 * khi) * NCLS + n] = accFC[r] + bb;
  }
}

extern "C" void kernel_launch(void* const* d_in, const int* in_sizes, int n_in,
                              void* d_out, int out_size, void* d_ws, size_t ws_size,
                              hipStream_t stream) {
  (void)in_sizes; (void)n_in; (void)d_ws; (void)ws_size; (void)out_size;
  const float* x     = (const float*)d_in[0];
  const float* W_ih1 = (const float*)d_in[1];
  const float* W_hh1 = (const float*)d_in[2];
  const float* b_ih1 = (const float*)d_in[3];
  const float* b_hh1 = (const float*)d_in[4];
  const float* W_ih2 = (const float*)d_in[5];
  const float* W_hh2 = (const float*)d_in[6];
  const float* b_ih2 = (const float*)d_in[7];
  const float* b_hh2 = (const float*)d_in[8];
  const float* W_fc  = (const float*)d_in[9];
  const float* b_fc  = (const float*)d_in[10];
  float* out = (float*)d_out;

  // LDS: 64KB swizzled weight fragments + 48KB wave-private h-states + 768B
  const size_t smem = (size_t)((1 * 8 + 4 * 8 + 4 * 4 + 2 * 4) * 512
                               + WAVES * 16 * (H1 + H2)) * sizeof(__hip_bfloat16)
                      + (H1 + H2) * sizeof(float);
  const int grid = BATCH / ROWS_PER_BLK;  // 128 workgroups of 8 waves
  rnn2_fused_wmma<<<grid, BLK, smem, stream>>>(x, W_ih1, W_hh1, b_ih1, b_hh1,
                                               W_ih2, W_hh2, b_ih2, b_hh2,
                                               W_fc, b_fc, out);
}
